// Transformer_65584150610343
// MI455X (gfx1250) — matrix operand order repaired and checked by reading, not compiled, not run on hardware
//
#include <hip/hip_runtime.h>
#include <math.h>

typedef _Float16 half_t;
typedef __attribute__((ext_vector_type(16))) _Float16 v16h;
typedef __attribute__((ext_vector_type(8)))  float    v8f;

#define WMMA_F16(a, b, c) \
  __builtin_amdgcn_wmma_f32_16x16x32_f16(false, (a), false, (b), (short)0, (c), false, false)

// A-fragment K index for 16-bit 16x32 A: lanes 0-15 hold K {0-7,16-23}, lanes 16-31 hold K {8-15,24-31}
__device__ __forceinline__ int a_kidx(int e, int hid) {
  return (e & 7) + ((e >> 3) << 4) + (hid << 3);
}

// branch-free tanh-approx GELU (matches jax.nn.gelu approximate=True)
__device__ __forceinline__ float gelu_tanh(float x) {
  const float k0 = 0.7978845608028654f;   // sqrt(2/pi)
  const float k1 = 0.044715f;
  float t = k0 * (x + k1 * x * x * x);
  float e = __expf(2.0f * t);
  float th = 1.0f - 2.0f / (e + 1.0f);    // == tanh(t), finite at +-inf
  return 0.5f * x * (1.0f + th);
}

// CDNA5 async global->LDS copy (ASYNCcnt-tracked), 16 bytes per lane
__device__ __forceinline__ void async_b128(uint32_t lds_off, const half_t* g) {
  asm volatile("global_load_async_to_lds_b128 %0, %1, off"
               :: "v"(lds_off), "v"(g) : "memory");
}

union FragU {
  uint4 u[2];
  v16h  v;
};

// ---------------------------------------------------------------------------
// Tiled WMMA GEMM: C = epilogue(A[MxK] @ Bt[NxK]^T + bias)
// B is pre-transposed to [N][K] so both tiles are contiguous row copies.
// Block = 256 threads (8 waves). Block tile 128x128, K-step 32, double-
// buffered LDS filled with global_load_async_to_lds_b128 + s_wait_asynccnt.
// Wave tile 32x64 = 2x4 fragments -> 8 v_wmma per K-step.
// ---------------------------------------------------------------------------
#define BM 128
#define BN 128
#define BK 32
#define LDP 40                       // padded row: 40 halves = 80B (16B aligned)
#define BUFB (128 * LDP * 2)         // bytes per LDS buffer (10240)

__global__ __launch_bounds__(256) void gemm_f16_wmma(
    const half_t* __restrict__ A, const half_t* __restrict__ Bt,
    const float* __restrict__ bias, const float* __restrict__ residual,
    float* __restrict__ out_f32, half_t* __restrict__ out_f16,
    int M, int N, int K, int apply_gelu)
{
  __shared__ half_t As[2][BM][LDP];
  __shared__ half_t Bs[2][BN][LDP];

  const int tid  = threadIdx.x;
  const int lane = tid & 31;
  const int wid  = tid >> 5;
  const int hid  = (lane >> 4) & 1;
  const int nl   = lane & 15;
  const int wm   = (wid >> 1) * 32;   // wave row offset in block tile (0..96)
  const int wn   = (wid & 1)  * 64;   // wave col offset in block tile (0/64)

  const int m0 = blockIdx.y * BM;
  const int n0 = blockIdx.x * BN;

  v8f acc[2][4];
  #pragma unroll
  for (int i = 0; i < 2; ++i)
    #pragma unroll
    for (int j = 0; j < 4; ++j)
      #pragma unroll
      for (int r = 0; r < 8; ++r) acc[i][j][r] = 0.0f;

  // async staging: each thread copies 32B of A and 32B of B per tile
  const int ldr = tid >> 1;           // tile row 0..127
  const int ldc = (tid & 1) * 16;     // half offset 0/16 within 32-wide K
  const half_t* aRow = A  + (size_t)(m0 + ldr) * K + ldc;
  const half_t* bRow = Bt + (size_t)(n0 + ldr) * K + ldc;
  const uint32_t aLds = (uint32_t)(size_t)&As[0][ldr][ldc];
  const uint32_t bLds = (uint32_t)(size_t)&Bs[0][ldr][ldc];

  const int nk = K / BK;

  // prefetch tile 0 into buffer 0
  async_b128(aLds,      aRow);
  async_b128(aLds + 16, aRow + 8);
  async_b128(bLds,      bRow);
  async_b128(bLds + 16, bRow + 8);

  for (int it = 0; it < nk; ++it) {
    const int cur = it & 1;
    if (it + 1 < nk) {
      const uint32_t boff = (uint32_t)((cur ^ 1) * BUFB);
      const int kn = (it + 1) * BK;
      async_b128(aLds + boff,      aRow + kn);
      async_b128(aLds + boff + 16, aRow + kn + 8);
      async_b128(bLds + boff,      bRow + kn);
      async_b128(bLds + boff + 16, bRow + kn + 8);
      asm volatile("s_wait_asynccnt 0x4" ::: "memory");  // current tile landed
    } else {
      asm volatile("s_wait_asynccnt 0x0" ::: "memory");
    }
    __syncthreads();   // all waves' async data visible

    v16h af[2], bf[4];
    #pragma unroll
    for (int i = 0; i < 2; ++i) {
      const int mr = wm + i * 16 + nl;
      FragU t;
      t.u[0] = *(const uint4*)&As[cur][mr][hid * 8];        // K = 8*hid..+7
      t.u[1] = *(const uint4*)&As[cur][mr][16 + hid * 8];   // K = 16+8*hid..+7
      af[i] = t.v;
    }
    #pragma unroll
    for (int j = 0; j < 4; ++j) {
      const int nc = wn + j * 16 + nl;
      FragU t;
      t.u[0] = *(const uint4*)&Bs[cur][nc][hid * 8];        // K = 8*hid..+7
      t.u[1] = *(const uint4*)&Bs[cur][nc][16 + hid * 8];   // K = 16+8*hid..+7
      bf[j] = t.v;
    }
    #pragma unroll
    for (int i = 0; i < 2; ++i)
      #pragma unroll
      for (int j = 0; j < 4; ++j)
        acc[i][j] = WMMA_F16(af[i], bf[j], acc[i][j]);

    __syncthreads();   // everyone done reading buf[cur] before overwrite
  }

  // epilogue: C layout -> lane holds col (wn+j*16+nl), rows r+8*hid
  #pragma unroll
  for (int i = 0; i < 2; ++i) {
    #pragma unroll
    for (int j = 0; j < 4; ++j) {
      const int col = n0 + wn + j * 16 + nl;
      const float bv = bias[col];
      #pragma unroll
      for (int r = 0; r < 8; ++r) {
        const int row = m0 + wm + i * 16 + r + hid * 8;
        float v = acc[i][j][r] + bv;
        if (apply_gelu) v = gelu_tanh(v);
        const size_t idx = (size_t)row * N + col;
        if (residual) v += residual[idx];
        if (out_f32) out_f32[idx] = v;
        if (out_f16) out_f16[idx] = (half_t)v;
      }
    }
  }
}

// ---------------------------------------------------------------------------
// Flash-style attention: one wave handles 16 query rows of one (b,h),
// online softmax over the key dimension in 32-key steps, all matmuls WMMA.
// qkv layout: [B, T, 3E]; q at col h*64, k at E + h*64, v at 2E + h*64.
// ---------------------------------------------------------------------------
__global__ __launch_bounds__(128) void attention_wmma(
    const half_t* __restrict__ qkv, half_t* __restrict__ out,
    int T, int E, int H)
{
  const int HD = 64;
  __shared__ half_t plds[4][16][34];  // per-wave P transpose staging

  const int tid  = threadIdx.x;
  const int lane = tid & 31;
  const int wid  = tid >> 5;
  const int hid  = lane >> 4;
  const int nl   = lane & 15;

  const int bh = blockIdx.y;
  const int b  = bh / H;
  const int h  = bh % H;
  const int qt = blockIdx.x * 4 + wid;
  const int qrow0 = qt * 16;

  const size_t rs = (size_t)3 * E;
  const half_t* qb = qkv + (size_t)b * T * rs + (size_t)h * HD;
  const half_t* kb = qb + E;
  const half_t* vb = qb + 2 * E;

  v16h qa[2];
  #pragma unroll
  for (int dc = 0; dc < 2; ++dc)
    #pragma unroll
    for (int e = 0; e < 16; ++e) {
      float qv = (float)qb[(size_t)(qrow0 + nl) * rs + dc * 32 + a_kidx(e, hid)];
      qa[dc][e] = (half_t)(qv * 0.125f);   // 1/sqrt(64)
    }

  float mrow[8], lrow[8];
  v8f o[4];
  #pragma unroll
  for (int j = 0; j < 8; ++j) { mrow[j] = -3.0e38f; lrow[j] = 0.0f; }
  #pragma unroll
  for (int t = 0; t < 4; ++t)
    #pragma unroll
    for (int r = 0; r < 8; ++r) o[t][r] = 0.0f;

  for (int kt = 0; kt < T / 32; ++kt) {
    v8f s[2];
    #pragma unroll
    for (int st = 0; st < 2; ++st) {
      #pragma unroll
      for (int r = 0; r < 8; ++r) s[st][r] = 0.0f;
      const int kr0 = kt * 32 + st * 16;
      #pragma unroll
      for (int dc = 0; dc < 2; ++dc) {
        v16h kf;  // B-frag: lane = key column, d in a_kidx order per lane half
        #pragma unroll
        for (int e = 0; e < 16; ++e)
          kf[e] = kb[(size_t)(kr0 + nl) * rs + dc * 32 + a_kidx(e, hid)];
        s[st] = WMMA_F16(qa[dc], kf, s[st]);
      }
    }

    float mn[8], al[8];
    #pragma unroll
    for (int j = 0; j < 8; ++j) {
      float mx = fmaxf(s[0][j], s[1][j]);
      #pragma unroll
      for (int d = 1; d < 16; d <<= 1) mx = fmaxf(mx, __shfl_xor(mx, d, 32));
      mn[j] = fmaxf(mrow[j], mx);
      al[j] = __expf(mrow[j] - mn[j]);
      mrow[j] = mn[j];
    }
    float p0[8], p1[8];
    #pragma unroll
    for (int j = 0; j < 8; ++j) {
      p0[j] = __expf(s[0][j] - mn[j]);
      p1[j] = __expf(s[1][j] - mn[j]);
      float sum = p0[j] + p1[j];
      #pragma unroll
      for (int d = 1; d < 16; d <<= 1) sum += __shfl_xor(sum, d, 32);
      lrow[j] = lrow[j] * al[j] + sum;
      #pragma unroll
      for (int t = 0; t < 4; ++t) o[t][j] *= al[j];
    }

    // transpose P from C-layout to A-layout through wave-private LDS
    #pragma unroll
    for (int j = 0; j < 8; ++j) {
      plds[wid][j + hid * 8][nl]      = (half_t)p0[j];
      plds[wid][j + hid * 8][16 + nl] = (half_t)p1[j];
    }
    asm volatile("s_wait_dscnt 0x0" ::: "memory");   // LDS RAW within wave
    v16h pa;
    #pragma unroll
    for (int e = 0; e < 16; ++e) pa[e] = plds[wid][nl][a_kidx(e, hid)];

    #pragma unroll
    for (int t = 0; t < 4; ++t) {
      v16h vf;  // B-frag: lane = d column, 16 key rows per lane half
      #pragma unroll
      for (int e = 0; e < 16; ++e)
        vf[e] = vb[(size_t)(kt * 32 + a_kidx(e, hid)) * rs + t * 16 + nl];
      o[t] = WMMA_F16(pa, vf, o[t]);
    }
  }

  #pragma unroll
  for (int t = 0; t < 4; ++t)
    #pragma unroll
    for (int j = 0; j < 8; ++j) {
      const int trow = qrow0 + j + hid * 8;
      out[((size_t)b * T + trow) * E + h * HD + t * 16 + nl] = (half_t)(o[t][j] / lrow[j]);
    }
}

// ---------------------------------------------------------------------------
// fp32 [K][N] -> f16 [N][K] tiled transpose-convert (coalesced both sides)
// ---------------------------------------------------------------------------
__global__ __launch_bounds__(256) void cvt_transpose_f16(
    const float* __restrict__ in, half_t* __restrict__ out, int K, int N)
{
  __shared__ half_t tile[32][33];
  const int kb = blockIdx.y * 32;
  const int nb = blockIdx.x * 32;
  const int tx = threadIdx.x;   // 0..31
  const int ty = threadIdx.y;   // 0..7
  #pragma unroll
  for (int r = 0; r < 32; r += 8)
    tile[ty + r][tx] = (half_t)in[(size_t)(kb + ty + r) * N + (nb + tx)];
  __syncthreads();
  #pragma unroll
  for (int r = 0; r < 32; r += 8)
    out[(size_t)(nb + ty + r) * K + (kb + tx)] = tile[tx][ty + r];
}

__global__ void init_stream(const float* __restrict__ x, float* __restrict__ xf32,
                            half_t* __restrict__ xf16, size_t n) {
  size_t i = (size_t)blockIdx.x * blockDim.x + threadIdx.x;
  size_t stride = (size_t)gridDim.x * blockDim.x;
  for (; i < n; i += stride) {
    float v = x[i];
    xf32[i] = v;
    xf16[i] = (half_t)v;
  }
}

// ---------------------------------------------------------------------------
extern "C" void kernel_launch(void* const* d_in, const int* in_sizes, int n_in,
                              void* d_out, int out_size, void* d_ws, size_t ws_size,
                              hipStream_t stream)
{
  (void)in_sizes; (void)n_in; (void)out_size; (void)ws_size;
  const int Bn = 2, T = 2048, E = 1024, H = 16, L = 4;
  const int FF = 4 * E;
  const int M  = Bn * T;          // 4096 token rows

  const float* x     = (const float*)d_in[0];
  const float* Wqkv  = (const float*)d_in[1];
  const float* bqkv  = (const float*)d_in[2];
  const float* Wproj = (const float*)d_in[3];
  const float* bproj = (const float*)d_in[4];
  const float* W1    = (const float*)d_in[5];
  const float* b1    = (const float*)d_in[6];
  const float* W2    = (const float*)d_in[7];
  const float* b2    = (const float*)d_in[8];

  char* p = (char*)d_ws;
  auto alloc = [&](size_t bytes) -> char* {
    char* r = p;
    p += (bytes + 255) & ~(size_t)255;
    return r;
  };
  float*  xf32   = (float*) alloc((size_t)M * E * 4);   // residual stream (fp32)
  half_t* xf16   = (half_t*)alloc((size_t)M * E * 2);   // residual stream (f16)
  half_t* qkvb   = (half_t*)alloc((size_t)M * 3 * E * 2);
  half_t* attb   = (half_t*)alloc((size_t)M * E * 2);
  half_t* h1b    = (half_t*)alloc((size_t)M * FF * 2);
  half_t* wqkvT  = (half_t*)alloc((size_t)E * 3 * E * 2);  // [3E][E]
  half_t* wprojT = (half_t*)alloc((size_t)E * E * 2);      // [E][E]
  half_t* w1T    = (half_t*)alloc((size_t)E * FF * 2);     // [FF][E]
  half_t* w2T    = (half_t*)alloc((size_t)FF * E * 2);     // [E][FF]

  init_stream<<<2048, 256, 0, stream>>>(x, xf32, xf16, (size_t)M * E);

  const dim3 tb(32, 8);
  for (int l = 0; l < L; ++l) {
    cvt_transpose_f16<<<dim3(3 * E / 32, E / 32), tb, 0, stream>>>(
        Wqkv + (size_t)l * E * 3 * E, wqkvT, E, 3 * E);
    cvt_transpose_f16<<<dim3(E / 32, E / 32), tb, 0, stream>>>(
        Wproj + (size_t)l * E * E, wprojT, E, E);
    cvt_transpose_f16<<<dim3(FF / 32, E / 32), tb, 0, stream>>>(
        W1 + (size_t)l * E * FF, w1T, E, FF);
    cvt_transpose_f16<<<dim3(E / 32, FF / 32), tb, 0, stream>>>(
        W2 + (size_t)l * FF * E, w2T, FF, E);

    // qkv = x @ Wqkv + bqkv
    gemm_f16_wmma<<<dim3(3 * E / BN, M / BM), 256, 0, stream>>>(
        xf16, wqkvT, bqkv + (size_t)l * 3 * E, nullptr, nullptr, qkvb,
        M, 3 * E, E, 0);

    // y = softmax(q k^T / sqrt(hd)) v
    attention_wmma<<<dim3(T / 64, Bn * H), 128, 0, stream>>>(qkvb, attb, T, E, H);

    // x = x + y @ Wproj + bproj
    gemm_f16_wmma<<<dim3(E / BN, M / BM), 256, 0, stream>>>(
        attb, wprojT, bproj + (size_t)l * E, xf32, xf32, xf16,
        M, E, E, 0);

    // h = gelu(x @ W1 + b1)
    gemm_f16_wmma<<<dim3(FF / BN, M / BM), 256, 0, stream>>>(
        xf16, w1T, b1 + (size_t)l * FF, nullptr, nullptr, h1b,
        M, FF, E, 1);

    // x = x + h @ W2 + b2   (last layer writes fp32 straight to d_out)
    float* of32 = (l == L - 1) ? (float*)d_out : xf32;
    gemm_f16_wmma<<<dim3(E / BN, M / BM), 256, 0, stream>>>(
        h1b, w2T, b2 + (size_t)l * E, xf32, of32, xf16,
        M, E, FF, 0);
  }
}
